// MoELayer_44135083934081
// MI455X (gfx1250) — compile-verified, matrix operand order checked by reading, not run on hardware
//
#include <hip/hip_runtime.h>
#include <hip/hip_bf16.h>
#include <stdint.h>

#define Bb 4
#define Ss 2048
#define Dd 2048
#define FFf 8192
#define Ee 2
#define Tt (Bb * Ss)   // 8192 tokens

typedef __attribute__((ext_vector_type(16))) __bf16 v16bf;
typedef __attribute__((ext_vector_type(8)))  float  v8f;
typedef __attribute__((ext_vector_type(4)))  unsigned int u32x4;
typedef __attribute__((ext_vector_type(8)))  int i32x8;
typedef __attribute__((ext_vector_type(4)))  int i32x4;

union Frag { v16bf v; unsigned int u[8]; };

__device__ __forceinline__ unsigned short f2bf(float f) {
  union { float f; unsigned int u; } c; c.f = f;
  unsigned int r = c.u + 0x7FFFu + ((c.u >> 16) & 1u);
  return (unsigned short)(r >> 16);
}
__device__ __forceinline__ unsigned int pack2bf(float lo, float hi) {
  return (unsigned int)f2bf(lo) | ((unsigned int)f2bf(hi) << 16);
}

// TDM: async-load a 128-row x 32-element (bf16) A tile into LDS.
// D# per CDNA5 ISA. LDS padding: 16 B after every
// 64 B stored -> row stride 80 B == uint[20] rows (matches fragment loads).
__device__ __forceinline__ void tdm_load_a_tile(unsigned lds_off,
                                                const unsigned short* gsrc,
                                                unsigned rowLen /*elems*/) {
  unsigned long long ga = (unsigned long long)(size_t)gsrc;
  u32x4 g0 = { 1u,                                   // count=1
               lds_off,                              // lds_addr
               (unsigned)ga,                         // global_addr[31:0]
               ((unsigned)((ga >> 32) & 0x1FFFFFFu)) | (2u << 30) }; // type=2
  i32x8 g1 = { (int)((1u << 16) | (1u << 20) | (3u << 22) | (3u << 25)),
               // data_size=2B | pad_enable | pad_interval=16DW | pad_amount=4DW
               (int)((rowLen & 0xFFFFu) << 16),      // tensor_dim0 lo
               (int)((rowLen >> 16) | ((unsigned)Tt << 16)),  // dim0 hi|dim1 lo
               (int)(32u << 16),                     // dim1 hi=0 | tile_dim0=32
               128,                                  // tile_dim1=128, tile_dim2=0
               (int)rowLen,                          // tensor_dim0_stride lo
               0, 0 };
  i32x4 gz4 = { 0, 0, 0, 0 };
  i32x8 gz8 = { 0, 0, 0, 0, 0, 0, 0, 0 };
  __builtin_amdgcn_tensor_load_to_lds(g0, g1, gz4, gz4, gz8, 0);
}

// ---------------- Kernel 1: router (logits -> softmax -> top-1 + slot) ------
__global__ __launch_bounds__(256) void moe_router(
    const float* __restrict__ x, const float* __restrict__ gw,
    int* __restrict__ sel, int* __restrict__ pos,
    float* __restrict__ topw, int* __restrict__ cnt) {
  int lane = threadIdx.x & 31;
  int wave = threadIdx.x >> 5;
  int t = blockIdx.x * 8 + wave;
  if (t >= Tt) return;
  const float4* xv = (const float4*)(x + (size_t)t * Dd);
  const float4* g0 = (const float4*)(gw);
  const float4* g1 = (const float4*)(gw + Dd);
  float a0 = 0.f, a1 = 0.f;
  for (int i = lane; i < Dd / 4; i += 32) {
    float4 xx = xv[i]; float4 w0 = g0[i]; float4 w1 = g1[i];
    a0 += xx.x * w0.x + xx.y * w0.y + xx.z * w0.z + xx.w * w0.w;
    a1 += xx.x * w1.x + xx.y * w1.y + xx.z * w1.z + xx.w * w1.w;
  }
  for (int off = 16; off > 0; off >>= 1) {
    a0 += __shfl_xor(a0, off, 32);
    a1 += __shfl_xor(a1, off, 32);
  }
  if (lane == 0) {
    int s = (a1 > a0) ? 1 : 0;
    float m = fmaxf(a0, a1);
    float e0 = __expf(a0 - m), e1 = __expf(a1 - m);
    float w = (s ? e1 : e0) / (e0 + e1);
    sel[t] = s;
    topw[t] = w;
    pos[t] = atomicAdd(&cnt[s], 1);
  }
}

// ---------------- Kernel 2: gather tokens -> expert-contiguous bf16 rows ----
__global__ __launch_bounds__(256) void moe_gather(
    const float* __restrict__ x, const int* __restrict__ sel,
    const int* __restrict__ pos, const float* __restrict__ topw,
    const int* __restrict__ cnt, unsigned short* __restrict__ xg,
    int* __restrict__ row_token, float* __restrict__ row_scale) {
  int t = blockIdx.x;
  int s = sel[t];
  int row = pos[t] + (s ? cnt[0] : 0);
  if (threadIdx.x == 0) { row_token[row] = t; row_scale[row] = topw[t]; }
  const float2* src = (const float2*)(x + (size_t)t * Dd);
  unsigned int* dst = (unsigned int*)(xg + (size_t)row * Dd);
  for (int i = threadIdx.x; i < Dd / 2; i += 256) {
    float2 v = src[i];
    dst[i] = pack2bf(v.x, v.y);
  }
}

// segment mapping helper: returns false if block has no work
__device__ __forceinline__ bool seg_map(int by, int c0, int& e, int& segBase,
                                        int& segCount, int& tileRow) {
  int t0 = (c0 + 127) >> 7;
  int c1 = Tt - c0;
  if (by < t0) { e = 0; segBase = 0; segCount = c0; tileRow = by << 7; return true; }
  int ty = by - t0;
  if ((ty << 7) >= c1) return false;
  e = 1; segBase = c0; segCount = c1; tileRow = ty << 7;
  return true;
}

// ---------------- Kernel 3: fused gate+up GEMM + SiLU -> h (bf16) -----------
// Block tile: 128 rows x 64 FF cols; 8 waves, each 16 rows x 64 cols.
// A panel streamed by TDM (double-buffered, TENSORcnt-paced); B converted
// fp32->bf16 on the VALU path.
__global__ __launch_bounds__(256) void moe_mlp1(
    const unsigned short* __restrict__ xg,
    const float* __restrict__ w_gate, const float* __restrict__ w_up,
    const int* __restrict__ cnt, unsigned short* __restrict__ h) {
  __shared__ unsigned int lds_a[2][128][20];  // two TDM-filled A buffers
  __shared__ unsigned int lds_bg[64][20];
  __shared__ unsigned int lds_bu[64][20];

  int e, segBase, segCount, tileRow;
  if (!seg_map(blockIdx.y, cnt[0], e, segBase, segCount, tileRow)) return;

  int f0 = blockIdx.x << 6;                 // 64 FF columns per block
  const float* wg = w_gate + (size_t)e * FFf * Dd;
  const float* wu = w_up   + (size_t)e * FFf * Dd;

  int tid = threadIdx.x;
  int lane = tid & 31, wave = tid >> 5;

  v8f accg[4] = {};
  v8f accu[4] = {};

  const unsigned short* abase = xg + (size_t)(segBase + tileRow) * Dd;
  unsigned ldsA[2] = { (unsigned)(size_t)&lds_a[0][0][0],
                       (unsigned)(size_t)&lds_a[1][0][0] };

  // B staging: each thread covers 4 consecutive K-pairs (8 floats) of one row
  int brow = tid >> 2;        // 0..63
  int bcol = (tid & 3) << 2;  // K-pair index: 0,4,8,12

  int m = (wave << 4) + (lane & 15);
  int pbase = (lane >> 4) << 2;

  if (wave == 0) tdm_load_a_tile(ldsA[0], abase, Dd);

  int cur = 0;
  for (int k0 = 0; k0 < Dd; k0 += 32) {
    const float4* wgp = (const float4*)(wg + (size_t)(f0 + brow) * Dd + k0 + bcol * 2);
    const float4* wup = (const float4*)(wu + (size_t)(f0 + brow) * Dd + k0 + bcol * 2);
    float4 g4a = wgp[0], g4b = wgp[1];
    float4 u4a = wup[0], u4b = wup[1];

    __syncthreads();                         // prior reads of lds_a[cur^1] done
    if (wave == 0) {
      int kn = (k0 + 32 < Dd) ? (k0 + 32) : 0;   // wrap: keeps wait imm const
      tdm_load_a_tile(ldsA[cur ^ 1], abase + kn, Dd);
    }
    uint4 gp = make_uint4(pack2bf(g4a.x, g4a.y), pack2bf(g4a.z, g4a.w),
                          pack2bf(g4b.x, g4b.y), pack2bf(g4b.z, g4b.w));
    uint4 up = make_uint4(pack2bf(u4a.x, u4a.y), pack2bf(u4a.z, u4a.w),
                          pack2bf(u4b.x, u4b.y), pack2bf(u4b.z, u4b.w));
    *(uint4*)&lds_bg[brow][bcol] = gp;
    *(uint4*)&lds_bu[brow][bcol] = up;
    __builtin_amdgcn_s_wait_tensorcnt(1);    // oldest (== buffer cur) complete
    __syncthreads();

    Frag a;
#pragma unroll
    for (int p = 0; p < 8; ++p) {
      int pi = pbase + (p < 4 ? p : p + 4);
      a.u[p] = lds_a[cur][m][pi];
    }
#pragma unroll
    for (int nt = 0; nt < 4; ++nt) {
      Frag bg, bu;
      int n = (nt << 4) + (lane & 15);
#pragma unroll
      for (int p = 0; p < 8; ++p) {
        int pi = pbase + (p < 4 ? p : p + 4);
        bg.u[p] = lds_bg[n][pi];
        bu.u[p] = lds_bu[n][pi];
      }
      accg[nt] = __builtin_amdgcn_wmma_f32_16x16x32_bf16(
          false, a.v, false, bg.v, (short)0, accg[nt], false, false);
      accu[nt] = __builtin_amdgcn_wmma_f32_16x16x32_bf16(
          false, a.v, false, bu.v, (short)0, accu[nt], false, false);
    }
    cur ^= 1;
  }

  int mrow0 = (wave << 4) + ((lane >> 4) << 3);
#pragma unroll
  for (int v = 0; v < 8; ++v) {
    int rl = tileRow + mrow0 + v;
    if (rl < segCount) {
      unsigned short* hp =
          h + (size_t)(segBase + rl) * FFf + f0 + (lane & 15);
#pragma unroll
      for (int nt = 0; nt < 4; ++nt) {
        float g = accg[nt][v], u = accu[nt][v];
        float sig = __builtin_amdgcn_rcpf(1.f + __expf(-g));
        hp[nt << 4] = f2bf(g * sig * u);    // silu(g)*u
      }
    }
  }
}

// ---------------- Kernel 4: down GEMM + routed-weight scatter to out --------
// Block tile: 128 rows x 64 D cols; A (h) streamed by TDM.
__global__ __launch_bounds__(256) void moe_mlp2(
    const unsigned short* __restrict__ h, const float* __restrict__ w_down,
    const int* __restrict__ cnt, const int* __restrict__ row_token,
    const float* __restrict__ row_scale, float* __restrict__ out) {
  __shared__ unsigned int lds_a[2][128][20];
  __shared__ unsigned int lds_b[64][20];

  int e, segBase, segCount, tileRow;
  if (!seg_map(blockIdx.y, cnt[0], e, segBase, segCount, tileRow)) return;

  int d0 = blockIdx.x << 6;                 // 64 D columns per block
  const float* wd = w_down + (size_t)e * Dd * FFf;

  int tid = threadIdx.x;
  int lane = tid & 31, wave = tid >> 5;

  v8f acc[4] = {};

  const unsigned short* abase = h + (size_t)(segBase + tileRow) * FFf;
  unsigned ldsA[2] = { (unsigned)(size_t)&lds_a[0][0][0],
                       (unsigned)(size_t)&lds_a[1][0][0] };

  int brow = tid >> 2;
  int bcol = (tid & 3) << 2;

  int m = (wave << 4) + (lane & 15);
  int pbase = (lane >> 4) << 2;

  if (wave == 0) tdm_load_a_tile(ldsA[0], abase, FFf);

  int cur = 0;
  for (int k0 = 0; k0 < FFf; k0 += 32) {
    const float4* wdp = (const float4*)(wd + (size_t)(d0 + brow) * FFf + k0 + bcol * 2);
    float4 d4a = wdp[0], d4b = wdp[1];

    __syncthreads();
    if (wave == 0) {
      int kn = (k0 + 32 < FFf) ? (k0 + 32) : 0;
      tdm_load_a_tile(ldsA[cur ^ 1], abase + kn, FFf);
    }
    uint4 dp = make_uint4(pack2bf(d4a.x, d4a.y), pack2bf(d4a.z, d4a.w),
                          pack2bf(d4b.x, d4b.y), pack2bf(d4b.z, d4b.w));
    *(uint4*)&lds_b[brow][bcol] = dp;
    __builtin_amdgcn_s_wait_tensorcnt(1);
    __syncthreads();

    Frag a;
#pragma unroll
    for (int p = 0; p < 8; ++p) {
      int pi = pbase + (p < 4 ? p : p + 4);
      a.u[p] = lds_a[cur][m][pi];
    }
#pragma unroll
    for (int nt = 0; nt < 4; ++nt) {
      Frag b;
      int n = (nt << 4) + (lane & 15);
#pragma unroll
      for (int p = 0; p < 8; ++p) {
        int pi = pbase + (p < 4 ? p : p + 4);
        b.u[p] = lds_b[n][pi];
      }
      acc[nt] = __builtin_amdgcn_wmma_f32_16x16x32_bf16(
          false, a.v, false, b.v, (short)0, acc[nt], false, false);
    }
    cur ^= 1;
  }

  int mrow0 = (wave << 4) + ((lane >> 4) << 3);
#pragma unroll
  for (int v = 0; v < 8; ++v) {
    int rl = tileRow + mrow0 + v;
    if (rl < segCount) {
      int row = segBase + rl;
      int tok = row_token[row];
      float sc = row_scale[row];
      float* orow = out + (size_t)tok * Dd + d0 + (lane & 15);
#pragma unroll
      for (int nt = 0; nt < 4; ++nt) orow[nt << 4] = sc * acc[nt][v];
    }
  }
}

// ---------------------------------------------------------------------------
extern "C" void kernel_launch(void* const* d_in, const int* in_sizes, int n_in,
                              void* d_out, int out_size, void* d_ws,
                              size_t ws_size, hipStream_t stream) {
  const float* x      = (const float*)d_in[0];
  const float* gate_w = (const float*)d_in[1];
  const float* w_gate = (const float*)d_in[2];
  const float* w_up   = (const float*)d_in[3];
  const float* w_down = (const float*)d_in[4];
  float* out = (float*)d_out;

  char* ws = (char*)d_ws;
  size_t off = 0;
  auto walloc = [&](size_t bytes) {
    off = (off + 255) & ~(size_t)255;
    char* p = ws + off;
    off += bytes;
    return p;
  };
  int*   cnt       = (int*)walloc(2 * sizeof(int));
  int*   sel       = (int*)walloc((size_t)Tt * sizeof(int));
  int*   pos       = (int*)walloc((size_t)Tt * sizeof(int));
  float* topw      = (float*)walloc((size_t)Tt * sizeof(float));
  int*   row_token = (int*)walloc((size_t)Tt * sizeof(int));
  float* row_scale = (float*)walloc((size_t)Tt * sizeof(float));
  // +128 slack rows so last-tile TDM reads stay inside the allocation
  unsigned short* xg   = (unsigned short*)walloc((size_t)(Tt + 128) * Dd * 2);
  unsigned short* hbuf = (unsigned short*)walloc((size_t)(Tt + 128) * FFf * 2);

  (void)hipMemsetAsync(cnt, 0, 2 * sizeof(int), stream);
  moe_router<<<Tt / 8, 256, 0, stream>>>(x, gate_w, sel, pos, topw, cnt);
  moe_gather<<<Tt, 256, 0, stream>>>(x, sel, pos, topw, cnt, xg, row_token,
                                     row_scale);
  // worst-case y-tiles: ceil(c0/128)+ceil(c1/128) <= Tt/128 + 1; use +2 slack
  dim3 g1(FFf / 64, Tt / 128 + 2);
  moe_mlp1<<<g1, 256, 0, stream>>>(xg, w_gate, w_up, cnt, hbuf);
  dim3 g2(Dd / 64, Tt / 128 + 2);
  moe_mlp2<<<g2, 256, 0, stream>>>(hbuf, w_down, cnt, row_token, row_scale,
                                   out);
}
